// LSTM_4028679324278
// MI455X (gfx1250) — matrix operand order repaired and checked by reading, not compiled, not run on hardware
//
#include <hip/hip_runtime.h>
#include <hip/hip_bf16.h>

typedef _Float16 v16h __attribute__((ext_vector_type(16)));
typedef _Float16 v8h  __attribute__((ext_vector_type(8)));
typedef float    v8f  __attribute__((ext_vector_type(8)));

// Problem constants
#define BB   128
#define TT   1024
#define SS   64
#define AA   8
#define HH   256
#define G4   1024          // 4*H gate width
#define NTIL 64            // G4 / 16 output tiles
#define KSLICE (NTIL * 32 * 16)   // halves per k-slice of packed weights (64 KB)

// Straight-line VALU transcendentals (v_exp_f32 + v_rcp_f32), correct saturation.
__device__ __forceinline__ float fast_sigmoid(float x) {
    return __builtin_amdgcn_rcpf(1.0f + __expf(-x));
}
__device__ __forceinline__ float fast_tanh(float x) {
    return 1.0f - 2.0f * __builtin_amdgcn_rcpf(1.0f + __expf(2.0f * x));
}

// Load a 16x32 f16 A-fragment (ISA 7.12.2 layout) from a row-major f16 buffer.
__device__ __forceinline__ v16h load_a_frag(const _Float16* p) {
    v8h lo = *(const v8h*)(p);
    v8h hi = *(const v8h*)(p + 16);
    v16h a;
#pragma unroll
    for (int i = 0; i < 8; ++i) { a[i] = lo[i]; a[i + 8] = hi[i]; }
    return a;
}

__device__ __forceinline__ v16h load_b_frag(const _Float16* __restrict__ Wp,
                                            int kk, int nt, int lane) {
    return *(const v16h*)(Wp + ((size_t)(kk * NTIL + nt) * 32 + lane) * 16);
}

// B fragment from a k-slice base pointer (slice chosen dynamically per step).
__device__ __forceinline__ v16h load_b_slice(const _Float16* wk, int nt, int lane) {
    return *(const v16h*)(wk + ((size_t)nt * 32 + lane) * 16);
}

__device__ __forceinline__ v8f zero8() {
    v8f z;
#pragma unroll
    for (int i = 0; i < 8; ++i) z[i] = 0.0f;
    return z;
}

// ---------------------------------------------------------------------------
// Recurrent LSTM layer (defined FIRST so the disasm window shows it).
// One block = 16 independent batch rows, full time loop; 512 threads = 16
// waves = 4 waves/SIMD (waves_per_eu(4,4) -> 256-VGPR budget).
// The k-slice order is ROTATED by t (ke = (kk+t)&7): mathematically the same
// accumulation, but the B-load addresses become time-dependent so LICM cannot
// hoist 256 VGPRs of weights out of the t-loop and spill them to scratch —
// the weights stream from L2 every step, triple-buffered against WMMA.
// ---------------------------------------------------------------------------
__global__ __attribute__((amdgpu_waves_per_eu(4, 4))) __launch_bounds__(512)
void lstm_rec_kernel(const float* __restrict__ xg,
                     const _Float16* __restrict__ Wp,
                     const int* __restrict__ length,
                     _Float16* __restrict__ zout,
                     float* __restrict__ out,
                     int final_layer) {
    __shared__ _Float16 h_lds[16 * 264];   // stride 264 halves: conflict-free A reads
    __shared__ float    c_lds[16 * 256];
    __shared__ float    g_lds[16 * 1028];  // stride 1028: avoids lane/lane+16 conflicts
    __shared__ int      lens[16];

    const int tid = threadIdx.x;
    const int lane = tid & 31, wave = tid >> 5;
    const int lane16 = lane & 15, hi16 = lane >> 4;
    const int nt0 = wave * 4;
    const int b0 = blockIdx.x * 16;
    const _Float16* ah = h_lds + lane16 * 264 + hi16 * 8;

    if (tid < 16) lens[tid] = length[b0 + tid];
    for (int i = tid; i < 16 * 264; i += 512) h_lds[i] = (_Float16)0.0f;
    for (int i = tid; i < 16 * 256; i += 512) c_lds[i] = 0.0f;
    __syncthreads();

    for (int t = 0; t < TT; ++t) {
        const int kr = t & 7;              // per-step k-slice rotation

        // ---- issue input-projection loads first (latency overlaps WMMA) ----
        float xv[4][8];
#pragma unroll
        for (int j = 0; j < 4; ++j) {
            int col = (nt0 + j) * 16 + lane16;
#pragma unroll
            for (int i = 0; i < 8; ++i) {
                int r = i + 8 * hi16;
                xv[j][i] = xg[((size_t)(b0 + r) * TT + t) * G4 + col];
            }
        }

        // ---- gates = h @ W_hh^T (f16 WMMA, f32 acc), 2-deep B pipeline ----
        v8f acc[4];
#pragma unroll
        for (int j = 0; j < 4; ++j) acc[j] = zero8();

        v16h b_buf[3][4];
        {
            const _Float16* wk0 = Wp + (size_t)kr * KSLICE;
            const _Float16* wk1 = Wp + (size_t)((kr + 1) & 7) * KSLICE;
#pragma unroll
            for (int j = 0; j < 4; ++j) b_buf[0][j] = load_b_slice(wk0, nt0 + j, lane);
#pragma unroll
            for (int j = 0; j < 4; ++j) b_buf[1][j] = load_b_slice(wk1, nt0 + j, lane);
        }
        v16h a_cur = load_a_frag(ah + kr * 32);

#pragma unroll
        for (int kk = 0; kk < 8; ++kk) {
            if (kk < 6) {
                const _Float16* wk2 = Wp + (size_t)((kr + kk + 2) & 7) * KSLICE;
#pragma unroll
                for (int j = 0; j < 4; ++j)
                    b_buf[(kk + 2) % 3][j] = load_b_slice(wk2, nt0 + j, lane);
            }
            v16h a_nxt;
            if (kk < 7) a_nxt = load_a_frag(ah + ((kr + kk + 1) & 7) * 32);
#pragma unroll
            for (int j = 0; j < 4; ++j)
                acc[j] = __builtin_amdgcn_wmma_f32_16x16x32_f16(
                    false, a_cur, false, b_buf[kk % 3][j], (short)0, acc[j], false, false);
            if (kk < 7) a_cur = a_nxt;
        }

        // ---- add precomputed input projection, stage gates in LDS ----
#pragma unroll
        for (int j = 0; j < 4; ++j) {
            int col = (nt0 + j) * 16 + lane16;
#pragma unroll
            for (int i = 0; i < 8; ++i) {
                int r = i + 8 * hi16;
                g_lds[r * 1028 + col] = acc[j][i] + xv[j][i];
            }
        }
        __syncthreads();

        // ---- pointwise LSTM cell update: 16*256 elements over 512 threads ----
#pragma unroll
        for (int e = 0; e < 8; ++e) {
            int idx = tid + e * 512;
            int r = idx >> 8, j = idx & 255;
            float gi = g_lds[r * 1028 + j];
            float gf = g_lds[r * 1028 + j + 256];
            float gg = g_lds[r * 1028 + j + 512];
            float go = g_lds[r * 1028 + j + 768];
            float c  = c_lds[r * 256 + j];
            c = fast_sigmoid(gf) * c + fast_sigmoid(gi) * fast_tanh(gg);
            float h = fast_sigmoid(go) * fast_tanh(c);
            c_lds[r * 256 + j] = c;
            h_lds[r * 264 + j] = (_Float16)h;
            if (!final_layer) {
                zout[((size_t)(b0 + r) * TT + t) * HH + j] = (_Float16)h;
            } else if (t == lens[r] - 1) {
                out[(size_t)(b0 + r) * HH + j] = h;
            }
        }
        __syncthreads();
    }
}

// ---------------------------------------------------------------------------
// Pack fp32 weight matrix W[N=1024][Kact] into f16 WMMA B-fragment layout.
// Zero-pads K beyond Kact (layer-0 K=72 padded to 96).
// ---------------------------------------------------------------------------
__global__ void pack_w_kernel(const float* __restrict__ W, int Kact,
                              _Float16* __restrict__ out, int total) {
    int tid = blockIdx.x * 256 + threadIdx.x;
    if (tid >= total) return;
    int d    = tid & 7;
    int lane = (tid >> 3) & 31;
    int nt   = (tid >> 8) & 63;
    int kt   = tid >> 14;
    int n  = nt * 16 + (lane & 15);
    int k0 = kt * 32 + (d >> 2) * 16 + (lane >> 4) * 8 + (d & 3) * 2;
    float f0 = (k0     < Kact) ? W[(size_t)n * Kact + k0]     : 0.0f;
    float f1 = (k0 + 1 < Kact) ? W[(size_t)n * Kact + k0 + 1] : 0.0f;
    out[(size_t)tid * 2]     = (_Float16)f0;
    out[(size_t)tid * 2 + 1] = (_Float16)f1;
}

__global__ void bias_kernel(const float* __restrict__ bi0, const float* __restrict__ bh0,
                            const float* __restrict__ bi1, const float* __restrict__ bh1,
                            float* __restrict__ bias0, float* __restrict__ bias1) {
    int n = blockIdx.x * 256 + threadIdx.x;
    if (n < G4) {
        bias0[n] = bi0[n] + bh0[n];
        bias1[n] = bi1[n] + bh1[n];
    }
}

// ---------------------------------------------------------------------------
// Layer-0 input projection: xg[r][0..1023] = concat(state,action)[r] @ W_ih0^T + bias
// K = 72 padded to 96 (3 k-tiles). Block = 16 rows x 1024 cols; 16 waves x 4 tiles.
// ---------------------------------------------------------------------------
__global__ __launch_bounds__(512) void xg0_kernel(const float* __restrict__ state,
                                                  const float* __restrict__ action,
                                                  const _Float16* __restrict__ Wp,
                                                  const float* __restrict__ bias,
                                                  float* __restrict__ xg) {
    __shared__ _Float16 x_lds[16 * 104];   // stride 104 halves: conflict-free A reads
    const int tid = threadIdx.x;
    const int lane = tid & 31, wave = tid >> 5;
    const int lane16 = lane & 15, hi16 = lane >> 4;
    const int nt0 = wave * 4;
    const size_t m0 = (size_t)blockIdx.x * 16;   // flattened row b*T + t

    for (int i = tid; i < 16 * 64; i += 512) {
        int rr = i >> 6, cc = i & 63;
        x_lds[rr * 104 + cc] = (_Float16)state[(m0 + rr) * 64 + cc];
    }
    for (int i = tid; i < 16 * 8; i += 512) {
        int rr = i >> 3, cc = i & 7;
        x_lds[rr * 104 + 64 + cc] = (_Float16)action[(m0 + rr) * 8 + cc];
    }
    for (int i = tid; i < 16 * 32; i += 512) {   // zero pad K = 72..103
        int rr = i >> 5, cc = i & 31;
        x_lds[rr * 104 + 72 + cc] = (_Float16)0.0f;
    }
    __syncthreads();

    v8f acc[4];
#pragma unroll
    for (int j = 0; j < 4; ++j) acc[j] = zero8();

    v16h a_cur = load_a_frag(x_lds + lane16 * 104 + hi16 * 8);
    v16h b_cur[4];
#pragma unroll
    for (int j = 0; j < 4; ++j) b_cur[j] = load_b_frag(Wp, 0, nt0 + j, lane);

#pragma unroll
    for (int kk = 0; kk < 3; ++kk) {
        v16h a_nxt;
        v16h b_nxt[4];
        if (kk < 2) {
            a_nxt = load_a_frag(x_lds + lane16 * 104 + (kk + 1) * 32 + hi16 * 8);
#pragma unroll
            for (int j = 0; j < 4; ++j) b_nxt[j] = load_b_frag(Wp, kk + 1, nt0 + j, lane);
        }
#pragma unroll
        for (int j = 0; j < 4; ++j)
            acc[j] = __builtin_amdgcn_wmma_f32_16x16x32_f16(
                false, a_cur, false, b_cur[j], (short)0, acc[j], false, false);
        if (kk < 2) {
            a_cur = a_nxt;
#pragma unroll
            for (int j = 0; j < 4; ++j) b_cur[j] = b_nxt[j];
        }
    }

#pragma unroll
    for (int j = 0; j < 4; ++j) {
        int col = (nt0 + j) * 16 + lane16;
        float bv = bias[col];
#pragma unroll
        for (int i = 0; i < 8; ++i) {
            int r = i + 8 * hi16;
            xg[(m0 + r) * G4 + col] = acc[j][i] + bv;
        }
    }
}

// ---------------------------------------------------------------------------
// Layer-1 input projection: xg[r] = z1[r] @ W_ih1^T + bias ; A from global f16 z1.
// K = 256 (8 k-tiles). 16 waves x 4 tiles, pipelined.
// ---------------------------------------------------------------------------
__global__ __launch_bounds__(512) void xg1_kernel(const _Float16* __restrict__ z1,
                                                  const _Float16* __restrict__ Wp,
                                                  const float* __restrict__ bias,
                                                  float* __restrict__ xg) {
    const int tid = threadIdx.x;
    const int lane = tid & 31, wave = tid >> 5;
    const int lane16 = lane & 15, hi16 = lane >> 4;
    const int nt0 = wave * 4;
    const size_t m0 = (size_t)blockIdx.x * 16;
    const _Float16* arow = z1 + (m0 + lane16) * HH + hi16 * 8;

    v8f acc[4];
#pragma unroll
    for (int j = 0; j < 4; ++j) acc[j] = zero8();

    v16h a_cur = load_a_frag(arow);
    v16h b_cur[4];
#pragma unroll
    for (int j = 0; j < 4; ++j) b_cur[j] = load_b_frag(Wp, 0, nt0 + j, lane);

#pragma unroll
    for (int kk = 0; kk < 8; ++kk) {
        v16h a_nxt;
        v16h b_nxt[4];
        if (kk < 7) {
            a_nxt = load_a_frag(arow + (kk + 1) * 32);
#pragma unroll
            for (int j = 0; j < 4; ++j) b_nxt[j] = load_b_frag(Wp, kk + 1, nt0 + j, lane);
        }
#pragma unroll
        for (int j = 0; j < 4; ++j)
            acc[j] = __builtin_amdgcn_wmma_f32_16x16x32_f16(
                false, a_cur, false, b_cur[j], (short)0, acc[j], false, false);
        if (kk < 7) {
            a_cur = a_nxt;
#pragma unroll
            for (int j = 0; j < 4; ++j) b_cur[j] = b_nxt[j];
        }
    }

#pragma unroll
    for (int j = 0; j < 4; ++j) {
        int col = (nt0 + j) * 16 + lane16;
        float bv = bias[col];
#pragma unroll
        for (int i = 0; i < 8; ++i) {
            int r = i + 8 * hi16;
            xg[(m0 + r) * G4 + col] = acc[j][i] + bv;
        }
    }
}

// ---------------------------------------------------------------------------
extern "C" void kernel_launch(void* const* d_in, const int* in_sizes, int n_in,
                              void* d_out, int out_size, void* d_ws, size_t ws_size,
                              hipStream_t stream) {
    const float* state  = (const float*)d_in[0];
    const float* action = (const float*)d_in[1];
    const int*   length = (const int*)d_in[2];
    const float* W_ih0  = (const float*)d_in[3];
    const float* W_hh0  = (const float*)d_in[4];
    const float* b_ih0  = (const float*)d_in[5];
    const float* b_hh0  = (const float*)d_in[6];
    const float* W_ih1  = (const float*)d_in[7];
    const float* W_hh1  = (const float*)d_in[8];
    const float* b_ih1  = (const float*)d_in[9];
    const float* b_hh1  = (const float*)d_in[10];
    float* out = (float*)d_out;

    const size_t BT = (size_t)BB * TT;
    char* ws = (char*)d_ws;
    size_t off = 0;
    float*     xg    = (float*)(ws + off);     off += BT * G4 * sizeof(float);      // 512 MB, reused per layer
    _Float16*  z1    = (_Float16*)(ws + off);  off += BT * HH * sizeof(_Float16);   // 64 MB
    _Float16*  wih0p = (_Float16*)(ws + off);  off += (size_t)3 * 64 * 32 * 16 * 2;
    _Float16*  whh0p = (_Float16*)(ws + off);  off += (size_t)8 * 64 * 32 * 16 * 2;
    _Float16*  wih1p = (_Float16*)(ws + off);  off += (size_t)8 * 64 * 32 * 16 * 2;
    _Float16*  whh1p = (_Float16*)(ws + off);  off += (size_t)8 * 64 * 32 * 16 * 2;
    float*     bias0 = (float*)(ws + off);     off += G4 * sizeof(float);
    float*     bias1 = (float*)(ws + off);     off += G4 * sizeof(float);

    // 1) pack weights into WMMA B-fragment layout (f16), fold biases
    pack_w_kernel<<<(3 * 16384) / 256, 256, 0, stream>>>(W_ih0, SS + AA, wih0p, 3 * 16384);
    pack_w_kernel<<<(8 * 16384) / 256, 256, 0, stream>>>(W_hh0, HH,      whh0p, 8 * 16384);
    pack_w_kernel<<<(8 * 16384) / 256, 256, 0, stream>>>(W_ih1, HH,      wih1p, 8 * 16384);
    pack_w_kernel<<<(8 * 16384) / 256, 256, 0, stream>>>(W_hh1, HH,      whh1p, 8 * 16384);
    bias_kernel<<<4, 256, 0, stream>>>(b_ih0, b_hh0, b_ih1, b_hh1, bias0, bias1);

    // 2) layer-0 input projection (parallel WMMA GEMM)
    xg0_kernel<<<BT / 16, 512, 0, stream>>>(state, action, wih0p, bias0, xg);

    // 3) layer-0 recurrence (8 blocks of 16 batch rows), writes z1 as f16
    lstm_rec_kernel<<<BB / 16, 512, 0, stream>>>(xg, whh0p, length, z1, out, 0);

    // 4) layer-1 input projection from z1 (xg buffer reused)
    xg1_kernel<<<BT / 16, 512, 0, stream>>>(z1, wih1p, bias1, xg);

    // 5) layer-1 recurrence, fused ragged gather into d_out
    lstm_rec_kernel<<<BB / 16, 512, 0, stream>>>(xg, whh1p, length, z1, out, 1);
}
